// LSTM_90271622627527
// MI455X (gfx1250) — matrix operand order repaired and checked by reading, not compiled, not run on hardware
//
#include <hip/hip_runtime.h>

// ---------------------------------------------------------------------------
// MI455X (gfx1250) LSTM: bf16 WMMA (16x16x32), persistent recurrent kernel
// with device-wide barrier, precomputed input projection, register-resident
// cell state, and a sched_barrier-pinned 3-stage software pipeline so WMMAs
// overlap the global loads of later K-chunks (no s_wait_loadcnt 0x0 stalls).
// ---------------------------------------------------------------------------

typedef __bf16 bf16;
typedef __attribute__((ext_vector_type(16))) __bf16 v16bf;
typedef __attribute__((ext_vector_type(8)))  __bf16 v8bf;
typedef __attribute__((ext_vector_type(4)))  __bf16 v4bf;
typedef __attribute__((ext_vector_type(8)))  float  v8f;

#define T_STEPS 256
#define BATCH   128
#define DIN     1024
#define HID     1024
#define G4      4096    // 4*HID
#define DH      2048    // DIN+HID

__device__ __forceinline__ float sigmoidf_(float x) {
    return 1.0f / (1.0f + __expf(-x));
}
__device__ __forceinline__ float tanhf_(float x) {
    float e = __expf(2.0f * x);
    return 1.0f - 2.0f / (e + 1.0f);   // correct limits at +/-inf
}

// ---------------------------------------------------------------------------
// WMMA fragment pipeline: one A tile (16x32) + four B tiles (32x16 each).
// 3-stage rotation, schedule pinned with sched_barrier so loads of chunk c+2
// are issued before the WMMAs of chunk c.
// ---------------------------------------------------------------------------
struct Frag {
    v16bf a;
    v16bf b[4];
};

__device__ __forceinline__ void load_frag(Frag& f, const bf16* __restrict__ A,
                                          const bf16* __restrict__ B0,
                                          int k0, int hi, size_t bstride) {
    union { v16bf v; v8bf h[2]; } a;
    a.h[0] = *(const v8bf*)(A + k0 + hi * 16);
    a.h[1] = *(const v8bf*)(A + k0 + hi * 16 + 8);
    f.a = a.v;
#pragma unroll
    for (int g = 0; g < 4; ++g)
        f.b[g] = *(const v16bf*)(B0 + g * bstride + k0);
}

__device__ __forceinline__ void mma4(v8f acc[4], const Frag& f) {
#pragma unroll
    for (int g = 0; g < 4; ++g)
        acc[g] = __builtin_amdgcn_wmma_f32_16x16x32_bf16(
            false, f.a, false, f.b[g], (short)0, acc[g], false, false);
}

// 16x64 (4x 16x16 tiles) GEMM panel over K with prefetch distance 2.
template <int K, size_t BSTRIDE>
__device__ __forceinline__ void gemm_16x64(v8f acc[4], const bf16* __restrict__ A,
                                           const bf16* __restrict__ B0, int hi) {
    constexpr int NC = K / 32;            // number of K-chunks
    Frag f[3];
    load_frag(f[0], A, B0, 0, hi, BSTRIDE);
    if (NC > 1) load_frag(f[1], A, B0, 32, hi, BSTRIDE);
    __builtin_amdgcn_sched_barrier(0);
#pragma unroll
    for (int c = 0; c < NC; ++c) {
        if (c + 2 < NC)
            load_frag(f[(c + 2) % 3], A, B0, (c + 2) * 32, hi, BSTRIDE);
        __builtin_amdgcn_sched_barrier(0);   // loads above, MMAs below
        mma4(acc, f[c % 3]);
        __builtin_amdgcn_sched_barrier(0);   // keep issue order across chunks
    }
}

// ---------------------------------------------------------------------------
// Utility kernels
// ---------------------------------------------------------------------------
__global__ void k_zero(float* __restrict__ p, int n) {
    int stride = gridDim.x * blockDim.x;
    for (int i = blockIdx.x * blockDim.x + threadIdx.x; i < n; i += stride)
        p[i] = 0.0f;
}

// float -> bf16, 4 elements per thread (n must be divisible by 4)
__global__ void k_cvt_bf16(bf16* __restrict__ dst, const float* __restrict__ src, int n4) {
    int stride = gridDim.x * blockDim.x;
    for (int i = blockIdx.x * blockDim.x + threadIdx.x; i < n4; i += stride) {
        float4 v = ((const float4*)src)[i];
        v4bf o;
        o[0] = (bf16)v.x; o[1] = (bf16)v.y; o[2] = (bf16)v.z; o[3] = (bf16)v.w;
        ((v4bf*)dst)[i] = o;
    }
}

// Split the four [H, D+H] fp32 gate weights into bf16 Wx4 [4H, D] and
// Wh4 [4H, H]; concatenate biases into b4 [4H] fp32.
__global__ void k_build_w4(bf16* __restrict__ Wx4, bf16* __restrict__ Wh4,
                           float* __restrict__ b4,
                           const float* __restrict__ Wf, const float* __restrict__ Wi,
                           const float* __restrict__ Wg, const float* __restrict__ Wo,
                           const float* __restrict__ bf_, const float* __restrict__ bi_,
                           const float* __restrict__ bg_, const float* __restrict__ bo_) {
    const float* Ws[4] = {Wf, Wi, Wg, Wo};
    const float* bs[4] = {bf_, bi_, bg_, bo_};
    const int total = G4 * DH;
    const int stride = gridDim.x * blockDim.x;
    for (int idx = blockIdx.x * blockDim.x + threadIdx.x; idx < total; idx += stride) {
        int row = idx >> 11;          // / DH
        int col = idx & (DH - 1);
        int g = row >> 10;            // gate
        int r = row & (HID - 1);
        float v = Ws[g][(size_t)r * DH + col];
        if (col < DIN) Wx4[(size_t)row * DIN + col] = (bf16)v;
        else           Wh4[(size_t)row * HID + (col - DIN)] = (bf16)v;
    }
    int gid = blockIdx.x * blockDim.x + threadIdx.x;
    if (gid < G4) b4[gid] = bs[gid >> 10][gid & (HID - 1)];
}

// ---------------------------------------------------------------------------
// Input projection: Zx[T*B, 4H] = Xbf[T*B, D] @ Wx4^T + b4   (bf16 WMMA)
// grid = 64 * (2048/8) = 16384 blocks, 256 threads; blocks of 8 waves share
// the B-panel (same n64) so weight rows hit the read-only WGP$.
// ---------------------------------------------------------------------------
__global__ void __launch_bounds__(256)
k_zx_gemm(const bf16* __restrict__ X, const bf16* __restrict__ Wx4,
          const float* __restrict__ b4, float* __restrict__ Zx) {
    const int lane = threadIdx.x & 31;
    const int w    = threadIdx.x >> 5;
    const int n64  = blockIdx.x & 63;                 // 0..63 : column group of 64
    const int mt   = (blockIdx.x >> 6) * 8 + w;       // 0..2047: 16-row tile
    const int l15  = lane & 15;
    const int hi   = lane >> 4;

    const bf16* A  = X   + (size_t)(mt * 16 + l15) * DIN;
    const bf16* B0 = Wx4 + (size_t)(n64 * 64 + l15) * DIN + hi * 16;

    v8f acc[4];
#pragma unroll
    for (int t = 0; t < 4; ++t) {
        float bv = b4[n64 * 64 + t * 16 + l15];
#pragma unroll
        for (int r = 0; r < 8; ++r) acc[t][r] = bv;
    }

    gemm_16x64<DIN, (size_t)16 * DIN>(acc, A, B0, hi);

#pragma unroll
    for (int t = 0; t < 4; ++t) {
        int col = n64 * 64 + t * 16 + l15;
#pragma unroll
        for (int r = 0; r < 8; ++r) {
            int row = mt * 16 + r + hi * 8;
            Zx[(size_t)row * G4 + col] = acc[t][r];
        }
    }
}

// ---------------------------------------------------------------------------
// Device-wide barrier (generation counter). bar[0]=arrive count, bar[1]=gen.
// ---------------------------------------------------------------------------
__device__ __forceinline__ void grid_sync(unsigned* bar, unsigned nblocks) {
    __threadfence();            // release my stores to agent scope
    __syncthreads();
    if (threadIdx.x == 0) {
        unsigned gen = __hip_atomic_load(&bar[1], __ATOMIC_RELAXED, __HIP_MEMORY_SCOPE_AGENT);
        unsigned prev = __hip_atomic_fetch_add(&bar[0], 1u, __ATOMIC_ACQ_REL, __HIP_MEMORY_SCOPE_AGENT);
        if (prev == nblocks - 1u) {
            __hip_atomic_store(&bar[0], 0u, __ATOMIC_RELAXED, __HIP_MEMORY_SCOPE_AGENT);
            __hip_atomic_store(&bar[1], gen + 1u, __ATOMIC_RELEASE, __HIP_MEMORY_SCOPE_AGENT);
        } else {
            while (__hip_atomic_load(&bar[1], __ATOMIC_ACQUIRE, __HIP_MEMORY_SCOPE_AGENT) == gen)
                __builtin_amdgcn_s_sleep(1);
        }
    }
    __syncthreads();
    __threadfence();            // acquire: invalidate read-only WGP$ for fresh h
}

// ---------------------------------------------------------------------------
// Persistent LSTM cell. grid = 64 blocks x 256 thr = 512 waves; wave (m,j)
// owns the 16x16 tile (batch rows m*16.., hidden cols j*16..) of all four
// gates forever:
//   - c state lives in registers for all 256 steps (never touches memory)
//   - Zx for step t+1 prefetched into registers during step t
//   - GEMM is the pipelined 16x64 panel (A reused across 4 gates)
// Wh4 (8MB bf16) and h (256KB) are L2-resident the whole time.
// ---------------------------------------------------------------------------
__global__ void __launch_bounds__(256)
k_lstm(const float* __restrict__ Zx, const bf16* __restrict__ Wh4,
       bf16* __restrict__ Hbf, unsigned* __restrict__ bar) {
    const int lane = threadIdx.x & 31;
    const int m    = threadIdx.x >> 5;   // 0..7  batch tile
    const int j    = blockIdx.x;         // 0..63 hidden tile
    const int l15  = lane & 15;
    const int hi   = lane >> 4;
    const int col  = j * 16 + l15;

    const bf16* A  = Hbf + (size_t)(m * 16 + l15) * HID;
    const bf16* B0 = Wh4 + (size_t)(j * 16 + l15) * HID + hi * 16;

    float creg[8];
    float zi[8], zf[8], zg[8], zo[8];
#pragma unroll
    for (int r = 0; r < 8; ++r) creg[r] = 0.0f;

    // prefetch Zx for t=0
#pragma unroll
    for (int r = 0; r < 8; ++r) {
        size_t zb = (size_t)(m * 16 + r + hi * 8) * G4 + col;
        zi[r] = Zx[zb];
        zf[r] = Zx[zb + HID];
        zg[r] = Zx[zb + 2 * HID];
        zo[r] = Zx[zb + 3 * HID];
    }

    for (int t = 0; t < T_STEPS; ++t) {
        v8f acc[4];
#pragma unroll
        for (int g = 0; g < 4; ++g)
#pragma unroll
            for (int r = 0; r < 8; ++r) acc[g][r] = 0.0f;

        gemm_16x64<HID, (size_t)HID * HID>(acc, A, B0, hi);

        // gate math; split order in reference: q0->i, q1->f, q2->g(tanh), q3->o
#pragma unroll
        for (int r = 0; r < 8; ++r) {
            int b = m * 16 + r + hi * 8;
            float it = sigmoidf_(acc[0][r] + zi[r]);
            float ft = sigmoidf_(acc[1][r] + zf[r]);
            float gt = tanhf_(acc[2][r] + zg[r]);
            float ot = sigmoidf_(acc[3][r] + zo[r]);
            float cv = ft * creg[r] + it * gt;
            creg[r] = cv;
            Hbf[(size_t)b * HID + col] = (bf16)(ot * tanhf_(cv));
        }

        // prefetch Zx for t+1 (completes during barrier + next GEMM);
        // issued AFTER this step's frag loads so in-order LOADcnt waits in the
        // GEMM never wait on these HBM loads.
        int tn = (t + 1 < T_STEPS) ? t + 1 : t;
        const float* zx = Zx + (size_t)tn * BATCH * G4;
#pragma unroll
        for (int r = 0; r < 8; ++r) {
            size_t zb = (size_t)(m * 16 + r + hi * 8) * G4 + col;
            zi[r] = zx[zb];
            zf[r] = zx[zb + HID];
            zg[r] = zx[zb + 2 * HID];
            zo[r] = zx[zb + 3 * HID];
        }

        grid_sync(bar, gridDim.x);
    }
}

// ---------------------------------------------------------------------------
// logits[128,1024] = hT @ Wout^T + bout   (bf16 WMMA). 128 wave-jobs.
// ---------------------------------------------------------------------------
__global__ void __launch_bounds__(256)
k_logits(const bf16* __restrict__ Hbf, const bf16* __restrict__ Wb,
         const float* __restrict__ bout, float* __restrict__ logits) {
    const int lane = threadIdx.x & 31;
    const int w    = threadIdx.x >> 5;
    const int wid  = blockIdx.x * 8 + w;     // 0..127
    const int mt   = wid >> 4;               // 0..7
    const int n64  = wid & 15;               // 0..15
    const int l15  = lane & 15;
    const int hi   = lane >> 4;

    const bf16* A  = Hbf + (size_t)(mt * 16 + l15) * HID;
    const bf16* B0 = Wb  + (size_t)(n64 * 64 + l15) * HID + hi * 16;

    v8f acc[4];
#pragma unroll
    for (int t = 0; t < 4; ++t) {
        float bv = bout[n64 * 64 + t * 16 + l15];
#pragma unroll
        for (int r = 0; r < 8; ++r) acc[t][r] = bv;
    }

    gemm_16x64<HID, (size_t)16 * HID>(acc, A, B0, hi);

#pragma unroll
    for (int t = 0; t < 4; ++t) {
        int col = n64 * 64 + t * 16 + l15;
#pragma unroll
        for (int r = 0; r < 8; ++r) {
            int row = mt * 16 + r + hi * 8;
            logits[(size_t)row * 1024 + col] = acc[t][r];
        }
    }
}

// ---------------------------------------------------------------------------
// log_softmax per row (128 rows x 1024).
// ---------------------------------------------------------------------------
__global__ void __launch_bounds__(256)
k_logsoftmax(const float* __restrict__ logits, float* __restrict__ out) {
    __shared__ float red[256];
    const int b = blockIdx.x;
    const float* row = logits + (size_t)b * 1024;

    float m = -3.402823466e38f;
    for (int i = threadIdx.x; i < 1024; i += 256) m = fmaxf(m, row[i]);
    red[threadIdx.x] = m;
    __syncthreads();
    for (int s = 128; s > 0; s >>= 1) {
        if (threadIdx.x < s) red[threadIdx.x] = fmaxf(red[threadIdx.x], red[threadIdx.x + s]);
        __syncthreads();
    }
    m = red[0];
    __syncthreads();

    float sum = 0.0f;
    for (int i = threadIdx.x; i < 1024; i += 256) sum += __expf(row[i] - m);
    red[threadIdx.x] = sum;
    __syncthreads();
    for (int s = 128; s > 0; s >>= 1) {
        if (threadIdx.x < s) red[threadIdx.x] += red[threadIdx.x + s];
        __syncthreads();
    }
    float lse = m + __logf(red[0]);
    for (int i = threadIdx.x; i < 1024; i += 256)
        out[(size_t)b * 1024 + i] = row[i] - lse;
}

// ---------------------------------------------------------------------------
// Host launcher
// ---------------------------------------------------------------------------
extern "C" void kernel_launch(void* const* d_in, const int* in_sizes, int n_in,
                              void* d_out, int out_size, void* d_ws, size_t ws_size,
                              hipStream_t stream) {
    (void)in_sizes; (void)n_in; (void)out_size; (void)ws_size;
    const float* x    = (const float*)d_in[0];
    const float* Wf   = (const float*)d_in[1];
    const float* bf_  = (const float*)d_in[2];
    const float* Wi   = (const float*)d_in[3];
    const float* bi_  = (const float*)d_in[4];
    const float* Wg   = (const float*)d_in[5];
    const float* bg_  = (const float*)d_in[6];
    const float* Wo   = (const float*)d_in[7];
    const float* bo_  = (const float*)d_in[8];
    const float* Wout = (const float*)d_in[9];
    const float* bout = (const float*)d_in[10];
    float* out = (float*)d_out;

    char* ws = (char*)d_ws;
    size_t off = 0;
    auto carve = [&](size_t bytes) -> char* {
        char* p = ws + off;
        off += (bytes + 255) & ~(size_t)255;
        return p;
    };
    // zeroed region first (h state + barrier are contiguous)
    bf16*     Hbf    = (bf16*) carve((size_t)BATCH * HID * sizeof(bf16));    // 256 KB
    unsigned* bar    = (unsigned*)carve(256);
    size_t zero_dwords = (size_t)(((char*)bar + 256) - (char*)Hbf) / 4;

    float* logits = (float*)carve((size_t)BATCH * 1024 * sizeof(float));
    bf16*  Xbf    = (bf16*) carve((size_t)T_STEPS * BATCH * DIN * sizeof(bf16)); // 64 MB
    bf16*  Wx4    = (bf16*) carve((size_t)G4 * DIN * sizeof(bf16));              // 8 MB
    bf16*  Wh4    = (bf16*) carve((size_t)G4 * HID * sizeof(bf16));              // 8 MB
    bf16*  Woutb  = (bf16*) carve((size_t)1024 * HID * sizeof(bf16));            // 2 MB
    float* b4     = (float*)carve((size_t)G4 * sizeof(float));
    float* Zx     = (float*)carve((size_t)T_STEPS * BATCH * G4 * sizeof(float)); // 512 MB

    k_zero<<<256, 256, 0, stream>>>((float*)Hbf, (int)zero_dwords);
    k_cvt_bf16<<<4096, 256, 0, stream>>>(Xbf, x, T_STEPS * BATCH * DIN / 4);
    k_cvt_bf16<<<1024, 256, 0, stream>>>(Woutb, Wout, 1024 * HID / 4);
    k_build_w4<<<4096, 256, 0, stream>>>(Wx4, Wh4, b4, Wf, Wi, Wg, Wo, bf_, bi_, bg_, bo_);
    k_zx_gemm<<<16384, 256, 0, stream>>>(Xbf, Wx4, b4, Zx);
    k_lstm<<<64, 256, 0, stream>>>(Zx, Wh4, Hbf, bar);
    k_logits<<<16, 256, 0, stream>>>(Hbf, Woutb, bout, logits);
    k_logsoftmax<<<BATCH, 256, 0, stream>>>(logits, out);
}
